// biLSTM_CRF_72095321030802
// MI455X (gfx1250) — matrix operand order repaired and checked by reading, not compiled, not run on hardware
//
#include <hip/hip_runtime.h>
#include <cstdint>
#include <cstddef>

// ---------------- problem dims ----------------
#define T_LEN 16384
#define D_IN  1024
#define H_DIR 512          // per-direction hidden
#define G4    2048         // 4*H gate rows
#define HID   1024         // 2*H
#define NCLS  16
#define NB    16           // recurrence blocks per direction

// gfx1250 async global->LDS copy path (ASYNCcnt) via inline asm
#define USE_ASYNC_LDS 1

typedef __attribute__((ext_vector_type(16))) __bf16 v16bf;
typedef __attribute__((ext_vector_type(8)))  __bf16 v8bf;
typedef __attribute__((ext_vector_type(8)))  float  v8f;

// ---------------- workspace layout (bytes) ----------------
static constexpr size_t NX  = (size_t)T_LEN * D_IN;     // X elements
static constexpr size_t NW  = (size_t)G4 * D_IN;        // W_ih elements per dir
static constexpr size_t XBF_OFF  = 0;
static constexpr size_t WFB_OFF  = XBF_OFF + NX * 2;
static constexpr size_t WBB_OFF  = WFB_OFF + NW * 2;
static constexpr size_t GF_OFF   = WBB_OFF + NW * 2;
static constexpr size_t GB_OFF   = GF_OFF + (size_t)T_LEN * G4 * 4;
static constexpr size_t LOUT_OFF = GB_OFF + (size_t)T_LEN * G4 * 4;
static constexpr size_t FEAT_OFF = LOUT_OFF + (size_t)T_LEN * HID * 4;
static constexpr size_t BP_OFF   = FEAT_OFF + (size_t)T_LEN * NCLS * 4;
static constexpr size_t HBUF_OFF = BP_OFF + (size_t)T_LEN * NCLS * 4;
static constexpr size_t CNT_OFF  = HBUF_OFF + (size_t)2 * 2 * H_DIR * 4;

__device__ __forceinline__ unsigned short f2bf(float f) {
    unsigned int u = __float_as_uint(f);
    unsigned int r = u + 0x7FFFu + ((u >> 16) & 1u);   // round-to-nearest-even
    return (unsigned short)(r >> 16);
}

// ============ kernel 0: f32 -> bf16 conversion + counter init ============
__global__ void __launch_bounds__(256)
convert_init_kernel(const float* __restrict__ X,
                    const float* __restrict__ Wf,
                    const float* __restrict__ Wb,
                    unsigned short* __restrict__ Xbf,
                    unsigned short* __restrict__ Wfb,
                    unsigned short* __restrict__ Wbb,
                    unsigned int* __restrict__ cnt) {
    size_t idx = (size_t)blockIdx.x * blockDim.x + threadIdx.x;
    if (idx == 0) { cnt[0] = 0u; cnt[64] = 0u; }   // per-direction step counters
    if (idx < NX) {
        Xbf[idx] = f2bf(X[idx]);
    } else if (idx < NX + NW) {
        size_t m = idx - NX;
        Wfb[m] = f2bf(Wf[m]);
    } else if (idx < NX + 2 * NW) {
        size_t m = idx - NX - NW;
        Wbb[m] = f2bf(Wb[m]);
    }
}

// ============ kernel 1: WMMA bf16 GEMM — precompute gates =================
// G_d[t][r] = sum_k X[t][k] * W_ih_d[r][k] + b_ih_d[r] + b_hh_d[r]
// Each wave: one 16-row M tile x four 16-col N tiles; K-loop over D in 32s,
// software-pipelined 2 deep so fragment loads overlap the WMMA pipe.
#define LOAD_FRAG(A0, A1, B, K)                                            \
    A0 = *(const v8bf*)(arow + (K));                                       \
    A1 = *(const v8bf*)(arow + (K) + 8);                                   \
    _Pragma("unroll")                                                      \
    for (int j = 0; j < 4; ++j) B[j] = *(const v16bf*)(brow[j] + (K));

#define COMPUTE_FRAG(A0, A1, B)                                            \
    {                                                                      \
        v16bf a;                                                           \
        _Pragma("unroll")                                                  \
        for (int e = 0; e < 8; ++e) { a[e] = A0[e]; a[e + 8] = A1[e]; }    \
        _Pragma("unroll")                                                  \
        for (int j = 0; j < 4; ++j)                                        \
            acc[j] = __builtin_amdgcn_wmma_f32_16x16x32_bf16(              \
                false, a, false, B[j], (short)0, acc[j], false, false);    \
    }

__global__ void __launch_bounds__(256)
gates_gemm_kernel(const unsigned short* __restrict__ Xbf,
                  const unsigned short* __restrict__ Wfb,
                  const unsigned short* __restrict__ Wbb,
                  const float* __restrict__ b_ih_f, const float* __restrict__ b_hh_f,
                  const float* __restrict__ b_ih_b, const float* __restrict__ b_hh_b,
                  float* __restrict__ Gf, float* __restrict__ Gb) {
    const int wave = threadIdx.x >> 5;
    const int lane = threadIdx.x & 31;
    const int lh   = lane & 15;
    const bool hiHalf = lane >= 16;

    const int MTILES  = T_LEN / 16;   // 1024
    const int NGROUPS = G4 / 64;      // 32 groups of 4 n-tiles
    size_t wg  = (size_t)blockIdx.x * 8 + wave;       // [0, 2*1024*32)
    int dir    = (int)(wg / ((size_t)MTILES * NGROUPS));
    size_t rem = wg % ((size_t)MTILES * NGROUPS);
    int mtile  = (int)(rem / NGROUPS);
    int ngroup = (int)(rem % NGROUPS);

    const unsigned short* W  = dir ? Wbb : Wfb;
    const float* bi = dir ? b_ih_b : b_ih_f;
    const float* bh = dir ? b_hh_b : b_hh_f;
    float* G = dir ? Gb : Gf;

    const int mbase  = mtile * 16;
    const int nbase0 = ngroup * 64;

    // A fragment source: 16x32 bf16, lane layout used here (same k order as B):
    //  lanes 0-15: M=lane,    K = 0..15 (contiguous)
    //  lanes16-31: M=lane-16, K = 16..31 (contiguous)
    const unsigned short* arow =
        Xbf + (size_t)(mbase + lh) * D_IN + (hiHalf ? 16 : 0);
    // B fragment source: 32x16 bf16 (K x N), N = W_ih row index:
    //  lanes 0-15: N=lane,    K = 0..15 (contiguous)
    //  lanes16-31: N=lane-16, K = 16..31 (contiguous)
    const unsigned short* brow[4];
#pragma unroll
    for (int j = 0; j < 4; ++j)
        brow[j] = W + (size_t)(nbase0 + j * 16 + lh) * D_IN + (hiHalf ? 16 : 0);

    v8f acc[4] = {};
    v8bf a0p, a1p, a0q, a1q;
    v16bf bp_[4], bq_[4];

    LOAD_FRAG(a0p, a1p, bp_, 0);                  // prologue: slice 0
    for (int kb = 0; kb < D_IN; kb += 64) {
        LOAD_FRAG(a0q, a1q, bq_, kb + 32);        // issue next slice's loads
        COMPUTE_FRAG(a0p, a1p, bp_);              // compute on current slice
        if (kb + 64 < D_IN) {
            __builtin_prefetch(arow + kb + 128, 0, 1);     // global_prefetch_b8
            __builtin_prefetch(brow[0] + kb + 128, 0, 1);
            LOAD_FRAG(a0p, a1p, bp_, kb + 64);
        }
        COMPUTE_FRAG(a0q, a1q, bq_);
    }

    // Epilogue: C/D layout — VGPR i: lanes 0-15 -> M=i, lanes 16-31 -> M=i+8
    const int rofs = hiHalf ? 8 : 0;
#pragma unroll
    for (int j = 0; j < 4; ++j) {
        int col = nbase0 + j * 16 + lh;
        float bias = bi[col] + bh[col];
#pragma unroll
        for (int i = 0; i < 8; ++i) {
            int row = mbase + i + rofs;
            G[(size_t)row * G4 + col] = acc[j][i] + bias;
        }
    }
}

// ============ kernel 2: persistent LDS-resident LSTM recurrence ===========
// 2 directions x NB blocks. Each block owns 32 hidden units (128 gate rows);
// its W_hh slice (128x512 f32 = 256KB) lives in LDS for the whole sequence.
// Per step: async global->LDS broadcast of h, wave-per-row dot products +
// pointwise gates; h double-buffered in global memory; release/acquire atomic
// counter barrier (with s_sleep backoff) across the NB blocks per direction.
__global__ void __launch_bounds__(256)
lstm_recurrence_kernel(const float* __restrict__ w_hh_f,
                       const float* __restrict__ w_hh_b,
                       const float* __restrict__ h0,
                       const float* __restrict__ c0,
                       const float* __restrict__ Gf,
                       const float* __restrict__ Gb,
                       float* __restrict__ lstm_out,
                       float* __restrict__ hbuf,        // [2 dir][2 buf][512]
                       unsigned int* __restrict__ cnt)  // [dir*64]
{
    extern __shared__ float smem[];
    float* wsl     = smem;                 // 128 x 512 weight slice
    float* h_sh    = wsl + 128 * 512;      // 512
    float* gate_sh = h_sh + 512;           // 128
    float* c_sh    = gate_sh + 128;        // 32

    const int tid  = threadIdx.x;
    const int dir  = blockIdx.x / NB;
    const int part = blockIdx.x % NB;
    const int u0   = part * (H_DIR / NB);  // first owned hidden unit (32 each)
    const float* Whh = dir ? w_hh_b : w_hh_f;
    const float* G   = dir ? Gb : Gf;
    unsigned int* c  = cnt + dir * 64;
    float* hb        = hbuf + (size_t)dir * 2 * H_DIR;

    // Load W_hh slice: local row lr = q*32 + j  <->  global row q*512 + u0 + j
    for (int e = tid; e < 128 * 512; e += 256) {
        int lr = e >> 9;
        int k  = e & 511;
        int q  = lr >> 5;
        int j  = lr & 31;
        wsl[e] = Whh[(size_t)(q * H_DIR + u0 + j) * H_DIR + k];
    }
    if (tid < 32) {
        c_sh[tid] = c0[dir * H_DIR + u0 + tid];
        hb[u0 + tid] = h0[dir * H_DIR + u0 + tid];   // buffer 0 = initial h
    }
    __threadfence();
    __syncthreads();
    if (tid == 0) {
        __hip_atomic_fetch_add(c, 1u, __ATOMIC_RELEASE, __HIP_MEMORY_SCOPE_AGENT);
        while (__hip_atomic_load(c, __ATOMIC_ACQUIRE, __HIP_MEMORY_SCOPE_AGENT) <
               (unsigned)NB) {
            __builtin_amdgcn_s_sleep(1);
        }
    }
    __syncthreads();

    const int wave = tid >> 5;
    const int lane = tid & 31;

#if USE_ASYNC_LDS
    // LDS byte offset of this thread's h_sh destination (8B per thread)
    const unsigned hsh_lds_off =
        (unsigned)(size_t)(__attribute__((address_space(3))) char*)(void*)(h_sh + 2 * tid);
#endif

    for (int t = 0; t < T_LEN; ++t) {
        const int time = dir ? (T_LEN - 1 - t) : t;
        const float* hprev = hb + (t & 1) * H_DIR;
#if USE_ASYNC_LDS
        {
            unsigned long long ga = (unsigned long long)(const void*)(hprev + 2 * tid);
            asm volatile("global_load_async_to_lds_b64 %0, %1, off"
                         :: "v"(hsh_lds_off), "v"(ga) : "memory");
            asm volatile("s_wait_asynccnt 0x0" ::: "memory");
        }
#else
        for (int k = tid; k < H_DIR; k += 256) h_sh[k] = hprev[k];
#endif
        __syncthreads();

        // 8 waves x 16 rows; lanes stride K (conflict-free LDS)
        for (int r = 0; r < 16; ++r) {
            const int lr = wave * 16 + r;
            const float* wr = wsl + (size_t)lr * H_DIR;
            float p = 0.f;
#pragma unroll 4
            for (int k = lane; k < H_DIR; k += 32) p += wr[k] * h_sh[k];
            p += __shfl_xor(p, 16, 32);
            p += __shfl_xor(p, 8, 32);
            p += __shfl_xor(p, 4, 32);
            p += __shfl_xor(p, 2, 32);
            p += __shfl_xor(p, 1, 32);
            if (lane == 0) {
                int q = lr >> 5, j = lr & 31;
                int grow = q * H_DIR + u0 + j;
                gate_sh[lr] = p + G[(size_t)time * G4 + grow];
            }
        }
        __syncthreads();

        if (tid < 32) {   // pointwise: i, f, g, o  (PyTorch gate order)
            float gi = gate_sh[tid];
            float gf = gate_sh[32 + tid];
            float gg = gate_sh[64 + tid];
            float go = gate_sh[96 + tid];
            float si = 1.f / (1.f + __expf(-gi));
            float sf = 1.f / (1.f + __expf(-gf));
            float so = 1.f / (1.f + __expf(-go));
            float cn = sf * c_sh[tid] + si * tanhf(gg);
            c_sh[tid] = cn;
            float hn = so * tanhf(cn);
            hb[((t + 1) & 1) * H_DIR + u0 + tid] = hn;
            lstm_out[(size_t)time * HID + dir * H_DIR + u0 + tid] = hn;
        }
        __threadfence();
        __syncthreads();
        if (tid == 0) {
            __hip_atomic_fetch_add(c, 1u, __ATOMIC_RELEASE, __HIP_MEMORY_SCOPE_AGENT);
            const unsigned target = (unsigned)(t + 2) * (unsigned)NB;
            while (__hip_atomic_load(c, __ATOMIC_ACQUIRE,
                                     __HIP_MEMORY_SCOPE_AGENT) < target) {
                __builtin_amdgcn_s_sleep(1);
            }
        }
        __syncthreads();
    }
}

// ============ kernel 3: linear head feats = lstm_out @ w_lin^T + b =========
__global__ void __launch_bounds__(256)
linear_kernel(const float* __restrict__ lstm_out,
              const float* __restrict__ w_lin,
              const float* __restrict__ b_lin,
              float* __restrict__ feats) {
    const int r  = threadIdx.x >> 4;
    const int cc = threadIdx.x & 15;
    const int row = blockIdx.x * 16 + r;
    const float4* x4 = (const float4*)(lstm_out + (size_t)row * HID);
    const float4* w4 = (const float4*)(w_lin + (size_t)cc * HID);
    float acc = b_lin[cc];
    for (int k = 0; k < HID / 4; ++k) {
        float4 x = x4[k], w = w4[k];
        acc += x.x * w.x + x.y * w.y + x.z * w.z + x.w * w.w;
    }
    feats[(size_t)row * NCLS + cc] = acc;
}

// ============ kernel 4: Viterbi decode (sequential, single block) ==========
__global__ void __launch_bounds__(256)
viterbi_kernel(const float* __restrict__ feats,
               const float* __restrict__ trans,
               int* __restrict__ bp,
               float* __restrict__ out) {
    __shared__ float fv[NCLS];
    __shared__ float tr[NCLS * NCLS];
    __shared__ float sc[NCLS][NCLS];
    const int tid = threadIdx.x;
    if (tid < NCLS * NCLS) tr[tid] = trans[tid];
    if (tid < NCLS) fv[tid] = 0.f;
    __syncthreads();
    const int n = tid >> 4, p = tid & 15;
    for (int t = 0; t < T_LEN; ++t) {
        sc[n][p] = fv[p] + tr[n * NCLS + p];
        __syncthreads();
        if (tid < NCLS) {
            float best = sc[tid][0];
            int bi = 0;
            for (int pp = 1; pp < NCLS; ++pp) {
                float v = sc[tid][pp];
                if (v > best) { best = v; bi = pp; }   // first-max tiebreak
            }
            bp[t * NCLS + tid] = bi;
            fv[tid] = best + feats[(size_t)t * NCLS + tid];
        }
        __syncthreads();
    }
    if (tid == 0) {
        int best = 0; float bv = fv[0];
        for (int i = 1; i < NCLS; ++i)
            if (fv[i] > bv) { bv = fv[i]; best = i; }
        out[0] = bv;                       // path_score
        out[1 + T_LEN] = (float)best;      // last path element
        int b = best;
        for (int t = T_LEN - 1; t >= 0; --t) {
            b = bp[t * NCLS + b];
            out[1 + t] = (float)b;
        }
    }
}

// ============================ launcher =====================================
extern "C" void kernel_launch(void* const* d_in, const int* in_sizes, int n_in,
                              void* d_out, int out_size, void* d_ws, size_t ws_size,
                              hipStream_t stream) {
    const float* X      = (const float*)d_in[0];
    const float* h0     = (const float*)d_in[1];
    const float* c0     = (const float*)d_in[2];
    const float* w_ih_f = (const float*)d_in[3];
    const float* w_hh_f = (const float*)d_in[4];
    const float* b_ih_f = (const float*)d_in[5];
    const float* b_hh_f = (const float*)d_in[6];
    const float* w_ih_b = (const float*)d_in[7];
    const float* w_hh_b = (const float*)d_in[8];
    const float* b_ih_b = (const float*)d_in[9];
    const float* b_hh_b = (const float*)d_in[10];
    const float* w_lin  = (const float*)d_in[11];
    const float* b_lin  = (const float*)d_in[12];
    const float* trans  = (const float*)d_in[13];

    char* ws = (char*)d_ws;
    unsigned short* Xbf = (unsigned short*)(ws + XBF_OFF);
    unsigned short* Wfb = (unsigned short*)(ws + WFB_OFF);
    unsigned short* Wbb = (unsigned short*)(ws + WBB_OFF);
    float* Gf    = (float*)(ws + GF_OFF);
    float* Gb    = (float*)(ws + GB_OFF);
    float* lout  = (float*)(ws + LOUT_OFF);
    float* feats = (float*)(ws + FEAT_OFF);
    int*   bp    = (int*)(ws + BP_OFF);
    float* hbuf  = (float*)(ws + HBUF_OFF);
    unsigned int* cnt = (unsigned int*)(ws + CNT_OFF);

    // 0) bf16 conversion + barrier-counter init
    {
        size_t total = NX + 2 * NW;
        int blocks = (int)((total + 255) / 256);
        convert_init_kernel<<<blocks, 256, 0, stream>>>(X, w_ih_f, w_ih_b,
                                                        Xbf, Wfb, Wbb, cnt);
    }
    // 1) WMMA gate GEMM: 2 dirs x 1024 Mtiles x 32 Ngroups waves = 8192 blocks
    gates_gemm_kernel<<<8192, 256, 0, stream>>>(Xbf, Wfb, Wbb,
                                                b_ih_f, b_hh_f, b_ih_b, b_hh_b,
                                                Gf, Gb);
    // 2) persistent recurrence: 32 blocks, ~259KB dynamic LDS each
    {
        size_t smem = (size_t)(128 * 512 + 512 + 128 + 32) * sizeof(float);
        (void)hipFuncSetAttribute((const void*)lstm_recurrence_kernel,
                                  hipFuncAttributeMaxDynamicSharedMemorySize,
                                  (int)smem);
        lstm_recurrence_kernel<<<2 * NB, 256, smem, stream>>>(
            w_hh_f, w_hh_b, h0, c0, Gf, Gb, lout, hbuf, cnt);
    }
    // 3) linear head
    linear_kernel<<<T_LEN / 16, 256, 0, stream>>>(lout, w_lin, b_lin, feats);
    // 4) Viterbi
    viterbi_kernel<<<1, 256, 0, stream>>>(feats, trans, bp, (float*)d_out);
}
